// AIMNetEmbed_46935402611398
// MI455X (gfx1250) — matrix operand order repaired and checked by reading, not compiled, not run on hardware
//
#include <hip/hip_runtime.h>
#include <hip/hip_bf16.h>
#include <math.h>

typedef __attribute__((ext_vector_type(16))) __bf16 v16bf;
typedef __attribute__((ext_vector_type(8)))  __bf16 v8bf;
typedef __attribute__((ext_vector_type(4)))  __bf16 v4bf;
typedef __attribute__((ext_vector_type(8)))  float  v8f;
typedef __attribute__((ext_vector_type(4)))  unsigned v4u;
typedef __attribute__((ext_vector_type(8)))  int    v8i;
typedef __attribute__((ext_vector_type(4)))  int    v4i;

// ---------------------------------------------------------------------------
// Problem constants
// ---------------------------------------------------------------------------
#define CB 8
#define CN 128
#define CA 64
#define CRR 32
#define CRA 16
#define CP 8128          // N*(N-1)/2
#define CDP 32           // d_pair
#define CH 512
#define CNE 256
#define CF (CRR * CA + CRA * CDP)  // 2560

// ---------------------------------------------------------------------------
// WMMA helpers (layouts per CDNA5 ISA §7.12.2, wave32)
// ---------------------------------------------------------------------------
__device__ __forceinline__ v8f wmma_bf16(v16bf a, v16bf b, v8f c) {
  return __builtin_amdgcn_wmma_f32_16x16x32_bf16(
      false, a, false, b, (short)0, c, false, false);
}

__device__ __forceinline__ v16bf cat8(v8bf lo, v8bf hi) {
  return __builtin_shufflevector(lo, hi, 0, 1, 2, 3, 4, 5, 6, 7,
                                 8, 9, 10, 11, 12, 13, 14, 15);
}

// A tile 16x32 from row-major bf16: two contiguous 16B chunks per lane
__device__ __forceinline__ v16bf load_a_bf16(const __bf16* Ablk, int lda) {
  int lane = threadIdx.x & 31;
  int half = lane >> 4, r = lane & 15;
  const __bf16* p = Ablk + r * lda + half * 16;
  return cat8(*(const v8bf*)p, *(const v8bf*)(p + 8));
}

// A tile 16x32 from row-major f32: two contiguous 32B chunks + cvt to bf16
__device__ __forceinline__ v16bf load_a_f32(const float* Ablk, long lda) {
  int lane = threadIdx.x & 31;
  int half = lane >> 4, r = lane & 15;
  const float* p = Ablk + (long)r * lda + half * 16;
  v8f c0 = *(const v8f*)p;
  v8f c1 = *(const v8f*)(p + 8);
  v16bf a;
#pragma unroll
  for (int i = 0; i < 8; ++i) {
    a[i] = (__bf16)c0[i];
    a[i + 8] = (__bf16)c1[i];
  }
  return a;
}

// B tile 32x16 from pre-swizzled panel: one contiguous 32B chunk per lane
__device__ __forceinline__ v16bf load_b_packed(const __bf16* tile) {
  return *(const v16bf*)(tile + (threadIdx.x & 31) * 16);
}

// C/D 16x16 f32: element v -> row v + 8*(lane>>4), col lane&15
__device__ __forceinline__ int c_row(int v) {
  return v + 8 * ((threadIdx.x & 31) >> 4);
}
__device__ __forceinline__ int c_col() { return threadIdx.x & 15; }

// branch-free gelu (tanh approx): tanh(y) = 1 - 2*rcp(1 + exp2(2*log2e*y))
__device__ __forceinline__ float gelu_tanh(float x) {
  const float k0 = 0.7978845608028654f;  // sqrt(2/pi)
  const float k1 = 0.044715f;
  float y = k0 * (x + k1 * x * x * x);
#if __has_builtin(__builtin_amdgcn_exp2f)
  float e = __builtin_amdgcn_exp2f(y * 2.8853900817779268f);  // exp(2y)
#else
  float e = exp2f(y * 2.8853900817779268f);
#endif
#if __has_builtin(__builtin_amdgcn_rcpf)
  float t = 1.0f - 2.0f * __builtin_amdgcn_rcpf(1.0f + e);
#else
  float t = 1.0f - 2.0f / (1.0f + e);
#endif
  return 0.5f * x * (1.0f + t);
}

// ---------------------------------------------------------------------------
// Kernel 0: pair index tables + afv passthrough to output tail
// ---------------------------------------------------------------------------
__global__ void k_init(const float* __restrict__ afv, float* __restrict__ out_afv,
                       int* __restrict__ ii, int* __restrict__ jj) {
  int t = blockIdx.x * blockDim.x + threadIdx.x;
  if (t < CB * CN * CA) out_afv[t] = afv[t];
  if (t < CN * CN) {
    int i = t / CN, j = t % CN;
    if (i < j) {
      int p = i * (2 * CN - i - 1) / 2 + (j - i - 1);
      ii[p] = i;
      jj[p] = j;
    }
  }
}

// ---------------------------------------------------------------------------
// Pack kernel: f32 KxN row-major -> bf16 WMMA-B swizzled panels.
//   dst[((kt*(N/16)+nt)*32 + lane)*16 + e] = src[(kt*32 + half*16 + e)*N + nt*16 + c]
//   lane = half*16 + c
// ---------------------------------------------------------------------------
__global__ void k_pack_b(const float* __restrict__ src, __bf16* __restrict__ dst,
                         int K, int N) {
  long total = (long)K * N;
  src += (long)blockIdx.y * total;
  dst += (long)blockIdx.y * total;
  for (long t = blockIdx.x * (long)blockDim.x + threadIdx.x; t < total;
       t += (long)gridDim.x * blockDim.x) {
    int k = (int)(t / N), n = (int)(t % N);
    int kt = k >> 5, half = (k >> 4) & 1, e = k & 15;
    int nt = n >> 4, c = n & 15;
    int lane = half * 16 + c;
    dst[(((long)kt * (N >> 4) + nt) * 32 + lane) * 16 + e] = (__bf16)src[t];
  }
}

// ---------------------------------------------------------------------------
// Kernel 1: pair MLP. 64 threads (2 waves), 32 rows/block, grid = B*P/32.
// Packed cw1+cw2 (contiguous 40KB in ws) staged into LDS once per block via
// the Tensor Data Mover; output written pre-swizzled for k_gav.
// ---------------------------------------------------------------------------
__global__ void k_pair_mlp(const float* __restrict__ afv,
                           const __bf16* __restrict__ w1p,  // packed 128x128 (+cw2 right after)
                           const float* __restrict__ cb1,
                           const __bf16* __restrict__ w2p,  // packed 128x32
                           const float* __restrict__ cb2,
                           const int* __restrict__ ii, const int* __restrict__ jj,
                           __bf16* __restrict__ pairp) {
  __shared__ __align__(64) __bf16 sW[20480];      // 40 KB: cw1p(16384) + cw2p(4096)
  __shared__ __align__(32) __bf16 sX[32 * 128];   //  8 KB
  __shared__ __align__(32) __bf16 sH[32 * 128];   //  8 KB

  int tid = threadIdx.x;  // 0..63
  int row0 = blockIdx.x * 32;

  // ---- stage weights into LDS (TDM: one 1-D tensor_load_to_lds per block) --
  if (tid < 32) {
#if __has_builtin(__builtin_amdgcn_tensor_load_to_lds)
    unsigned lds_off = (unsigned)(unsigned long long)(const void*)sW;
    unsigned long long gaddr = (unsigned long long)w1p;
    // D# group0: count=1 | lds_addr | global_addr[56:0] | type=2 (cdna5 §8.3)
    v4u g0 = {1u, lds_off, (unsigned)gaddr,
              (unsigned)((gaddr >> 32) & 0x1FFFFFFull) | (2u << 30)};
    // D# group1: data_size=1(2B); tensor_dim0=tile_dim0=stride=20480 elems (§8.4)
    v8i g1 = {(int)(1u << 16),          // mask=0, data_size=2B
              (int)(20480u << 16),      // tensor_dim0[15:0] @ bits[63:48]
              0,                        // tensor_dim0[31:16]=0, tensor_dim1=0
              (int)(20480u << 16),      // tile_dim0 @ bits[127:112]
              0,                        // tile_dim1=0, tile_dim2=0
              20480,                    // tensor_dim0_stride[31:0]
              0, 0};
    v4i z4 = {0, 0, 0, 0};
    v8i z8 = {0, 0, 0, 0, 0, 0, 0, 0};
    __builtin_amdgcn_tensor_load_to_lds(g0, g1, z4, z4, z8, 0);
    __builtin_amdgcn_s_wait_tensorcnt(0);
#else
    for (int idx = tid; idx < 20480; idx += 32) sW[idx] = w1p[idx];
#endif
  }

  // ---- build X = [ai+aj, ai*aj] (float4) ----
  for (int q = tid; q < 32 * 32; q += 64) {
    int rr = q >> 5;
    int c4 = (q & 31) * 4;
    int row = row0 + rr;
    int b = row / CP;
    int p = row - b * CP;
    int i = ii[p], j = jj[p];
    int a = c4 & 63;
    const float4 vi = *(const float4*)(afv + ((long)b * CN + i) * CA + a);
    const float4 vj = *(const float4*)(afv + ((long)b * CN + j) * CA + a);
    v4bf o;
    if (c4 < 64) {
      o[0] = (__bf16)(vi.x + vj.x); o[1] = (__bf16)(vi.y + vj.y);
      o[2] = (__bf16)(vi.z + vj.z); o[3] = (__bf16)(vi.w + vj.w);
    } else {
      o[0] = (__bf16)(vi.x * vj.x); o[1] = (__bf16)(vi.y * vj.y);
      o[2] = (__bf16)(vi.z * vj.z); o[3] = (__bf16)(vi.w * vj.w);
    }
    *(v4bf*)(sX + rr * 128 + c4) = o;
  }
  __syncthreads();

  int w = tid >> 5;  // wave 0..1
  int m0 = w * 16;

  // GEMM1: (16x128)@(128x128) + b1, gelu -> sH (B tiles from LDS)
  for (int nt = 0; nt < 8; ++nt) {
    v8f acc = {};
    for (int kt = 0; kt < 4; ++kt) {
      v16bf a = load_a_bf16(sX + m0 * 128 + kt * 32, 128);
      v16bf bm = load_b_packed(sW + (kt * 8 + nt) * 512);
      acc = wmma_bf16(a, bm, acc);
    }
#pragma unroll
    for (int v = 0; v < 8; ++v) {
      int r = m0 + c_row(v);
      int c = nt * 16 + c_col();
      sH[r * 128 + c] = (__bf16)gelu_tanh(acc[v] + cb1[c]);
    }
  }
  // each wave consumes only its own 16 rows of sH

  // GEMM2: (16x128)@(128x32) + b2 -> pairp (WMMA-B swizzled layout)
  for (int nt = 0; nt < 2; ++nt) {
    v8f acc = {};
    for (int kt = 0; kt < 4; ++kt) {
      v16bf a = load_a_bf16(sH + m0 * 128 + kt * 32, 128);
      v16bf bm = load_b_packed(sW + 16384 + (kt * 2 + nt) * 512);
      acc = wmma_bf16(a, bm, acc);
    }
#pragma unroll
    for (int v = 0; v < 8; ++v) {
      int rg = row0 + m0 + c_row(v);  // b*P + p
      int b = rg / CP;
      int p = rg - b * CP;
      int kt = p >> 5, halfp = (p >> 4) & 1, e = p & 15;
      int lane = halfp * 16 + c_col();
      float val = acc[v] + cb2[nt * 16 + c_col()];
      pairp[(long)b * (CP * CDP) + (((long)kt * 2 + nt) * 32 + lane) * 16 + e] =
          (__bf16)val;
    }
  }
}

// ---------------------------------------------------------------------------
// Kernel 2: grv[b,n] = gr[b,n] (32x128) @ afv[b] (128x64) -> X2[:, 0:2048]
// grid = B*N, 256 threads (8 waves: 2 M-tiles x 4 N-tiles)
// ---------------------------------------------------------------------------
__global__ void k_grv(const float* __restrict__ gr, const __bf16* __restrict__ afvp,
                      __bf16* __restrict__ X2) {
  int bn = blockIdx.x;
  int b = bn >> 7;
  int w = threadIdx.x >> 5;
  int mt = w >> 2, nt = w & 3;
  const float* Ablk = gr + ((long)bn * CRR + mt * 16) * CN;
  const __bf16* Bb = afvp + (long)b * (CN * CA);
  v8f acc = {};
  for (int kt = 0; kt < 4; ++kt) {
    v16bf a = load_a_f32(Ablk + kt * 32, CN);
    v16bf bm = load_b_packed(Bb + (kt * 4 + nt) * 512);
    acc = wmma_bf16(a, bm, acc);
  }
  __bf16* dst = X2 + (long)bn * CF;  // feature layout: r*64 + a
#pragma unroll
  for (int v = 0; v < 8; ++v) {
    int r = mt * 16 + c_row(v);
    int c = nt * 16 + c_col();
    dst[r * CA + c] = (__bf16)acc[v];
  }
}

// ---------------------------------------------------------------------------
// Kernel 3: gav per b: (2048 x 8128) @ afv_pair_b (8128 x 32) -> X2[:, 2048:]
// grid = B*128 M-tiles, 64 threads (2 waves = 2 N-tiles): ga streamed once.
// unroll-4 + global_prefetch to keep the 508 MiB HBM stream ahead of WMMA.
// ---------------------------------------------------------------------------
__global__ void k_gav(const float* __restrict__ ga, const __bf16* __restrict__ pairp,
                      __bf16* __restrict__ X2) {
  int b = blockIdx.x >> 7;
  int mt = blockIdx.x & 127;
  int w = threadIdx.x >> 5;  // N-tile 0/1
  const float* Ablk = ga + (long)b * (CN * CRA) * (long)CP + (long)(mt * 16) * CP;
  const __bf16* Bb = pairp + (long)b * (CP * CDP);
  int lane = threadIdx.x & 31;
  const float* pf = Ablk + (long)(lane & 15) * CP + (lane >> 4) * 8;
  v8f acc = {};
#pragma unroll 4
  for (int kt = 0; kt < CP / 32; ++kt) {  // 254
    __builtin_prefetch(pf + (kt + 16) * 32, 0, 1);  // ~2KB/row ahead
    v16bf a = load_a_f32(Ablk + kt * 32, CP);
    v16bf bm = load_b_packed(Bb + ((long)kt * 2 + w) * 512);
    acc = wmma_bf16(a, bm, acc);
  }
#pragma unroll
  for (int v = 0; v < 8; ++v) {
    int rg = mt * 16 + c_row(v);  // (n, a) row in [0, 2048)
    int n = rg >> 4, aa = rg & 15;
    int c = w * 16 + c_col();
    X2[((long)b * CN + n) * CF + CRR * CA + aa * CDP + c] = (__bf16)acc[v];
  }
}

// ---------------------------------------------------------------------------
// Kernel 4: aef = gelu(X2 @ ew1 + eb1) @ ew2 + eb2
// grid = 64 blocks (16 rows), 256 threads (8 waves); hidden staged in LDS
// ---------------------------------------------------------------------------
__global__ void k_embed_mlp(const __bf16* __restrict__ X2,
                            const __bf16* __restrict__ w1p,  // packed 2560x512
                            const float* __restrict__ eb1,
                            const __bf16* __restrict__ w2p,  // packed 512x256
                            const float* __restrict__ eb2, float* __restrict__ out) {
  __shared__ __align__(32) __bf16 sH[16 * CH];  // 16 KB
  int m0 = blockIdx.x * 16;
  int w = threadIdx.x >> 5;  // 0..7
  const __bf16* Ablk = X2 + (long)m0 * CF;

  // GEMM1: (16x2560)@(2560x512): 4 N-tiles per wave
  for (int nti = 0; nti < 4; ++nti) {
    int nt = w * 4 + nti;
    v8f acc = {};
    for (int kt = 0; kt < CF / 32; ++kt) {  // 80
      v16bf a = load_a_bf16(Ablk + kt * 32, CF);
      v16bf bm = load_b_packed(w1p + ((long)kt * 32 + nt) * 512);
      acc = wmma_bf16(a, bm, acc);
    }
#pragma unroll
    for (int v = 0; v < 8; ++v) {
      int r = c_row(v);
      int c = nt * 16 + c_col();
      sH[r * CH + c] = (__bf16)gelu_tanh(acc[v] + eb1[c]);
    }
  }
  __syncthreads();

  // GEMM2: (16x512)@(512x256): 2 N-tiles per wave
  for (int nti = 0; nti < 2; ++nti) {
    int nt = w * 2 + nti;
    v8f acc = {};
    for (int kt = 0; kt < CH / 32; ++kt) {  // 16
      v16bf a = load_a_bf16(sH + kt * 32, CH);
      v16bf bm = load_b_packed(w2p + (kt * 16 + nt) * 512);
      acc = wmma_bf16(a, bm, acc);
    }
#pragma unroll
    for (int v = 0; v < 8; ++v) {
      int r = m0 + c_row(v);
      int c = nt * 16 + c_col();
      out[(long)r * CNE + c] = acc[v] + eb2[c];
    }
  }
}

// ---------------------------------------------------------------------------
extern "C" void kernel_launch(void* const* d_in, const int* in_sizes, int n_in,
                              void* d_out, int out_size, void* d_ws, size_t ws_size,
                              hipStream_t stream) {
  const float* gr  = (const float*)d_in[0];
  const float* ga  = (const float*)d_in[1];
  const float* afv = (const float*)d_in[2];
  const float* cw1 = (const float*)d_in[3];
  const float* cb1 = (const float*)d_in[4];
  const float* cw2 = (const float*)d_in[5];
  const float* cb2 = (const float*)d_in[6];
  const float* ew1 = (const float*)d_in[7];
  const float* eb1 = (const float*)d_in[8];
  const float* ew2 = (const float*)d_in[9];
  const float* eb2 = (const float*)d_in[10];

  float* out_aef = (float*)d_out;                    // (B*N, 256)
  float* out_afv = out_aef + (size_t)CB * CN * CNE;  // (B*N, 64)

  // workspace carve-up (~12.5 MB); cw1p and cw2p are contiguous (one TDM copy)
  char* ws = (char*)d_ws;
  int*    w_ii    = (int*)(ws + 0);           // P ints
  int*    w_jj    = (int*)(ws + 32768);       // P ints
  __bf16* w_afvp  = (__bf16*)(ws + 65536);    // packed afv, B x (128x64)
  __bf16* w_pairp = (__bf16*)(ws + 196608);   // packed afv_pair, B x (Px32)
  __bf16* w_X2    = (__bf16*)(ws + 4358144);  // (B*N, 2560) row-major
  __bf16* w_cw1p  = (__bf16*)(ws + 9601024);  // packed 128x128 (32KB)
  __bf16* w_cw2p  = (__bf16*)(ws + 9633792);  // packed 128x32  (8KB, follows cw1p)
  __bf16* w_ew1p  = (__bf16*)(ws + 9641984);  // packed 2560x512
  __bf16* w_ew2p  = (__bf16*)(ws + 12263424); // packed 512x256

  // 0: indices + afv passthrough
  k_init<<<256, 256, 0, stream>>>(afv, out_afv, w_ii, w_jj);

  // pack all B-side operands into WMMA lane-major panels
  k_pack_b<<<dim3(64, 1), 256, 0, stream>>>(cw1, w_cw1p, 128, 128);
  k_pack_b<<<dim3(16, 1), 256, 0, stream>>>(cw2, w_cw2p, 128, 32);
  k_pack_b<<<dim3(2048, 1), 256, 0, stream>>>(ew1, w_ew1p, CF, CH);
  k_pack_b<<<dim3(512, 1), 256, 0, stream>>>(ew2, w_ew2p, CH, CNE);
  k_pack_b<<<dim3(32, CB), 256, 0, stream>>>(afv, w_afvp, CN, CA);

  // 1: pair MLP (B*P = 65024 rows, 32 per block; TDM stages weights to LDS)
  k_pair_mlp<<<(CB * CP) / 32, 64, 0, stream>>>(afv, w_cw1p, cb1, w_cw2p, cb2,
                                                w_ii, w_jj, w_pairp);

  // 2: grv einsum
  k_grv<<<CB * CN, 256, 0, stream>>>(gr, w_afvp, w_X2);

  // 3: gav einsum (streams ga once)
  k_gav<<<CB * 128, 64, 0, stream>>>(ga, w_pairp, w_X2);

  // 4: embedding MLP
  k_embed_mlp<<<(CB * CN) / 16, 256, 0, stream>>>(w_X2, w_ew1p, eb1, w_ew2p, eb2,
                                                  out_aef);
}
